// ViM_71949292142655
// MI455X (gfx1250) — compile-verified, matrix operand order checked by reading, not run on hardware
//
#include <hip/hip_runtime.h>
#include <hip/hip_bf16.h>

// ---------------------------------------------------------------------------
// ViT adapter MoE on gfx1250 (MI455X).
// 121 GFLOP vs ~240 MB traffic -> compute bound => bf16 WMMA, f32 accum.
// 3 GEMM stages (down / implicit-gemm 3x3 conv / up) on v_wmma_f32_16x16x32_bf16.
// bf16 tile movement uses GLOBAL_LOAD_ASYNC_TO_LDS_B128 (ASYNCcnt) with a
// double-buffered software pipeline in the conv kernel.
// ---------------------------------------------------------------------------

typedef __attribute__((ext_vector_type(16))) __bf16 v16bf;
typedef __attribute__((ext_vector_type(8)))  __bf16 v8bf;
typedef __attribute__((ext_vector_type(8)))  float  v8f;
typedef __attribute__((ext_vector_type(4)))  float  v4f;
typedef __attribute__((ext_vector_type(4)))  int    v4i;

#define DEV __device__ __forceinline__

// Problem constants
#define NTOK 577
#define BB   32
#define CC   1024
#define DD   128
#define EE   8
#define JJ   (NTOK * BB)           // 18464 GEMM rows
#define JTILES ((JJ + 127) / 128)  // 145
#define LS   40                    // LDS row stride in bf16 (80 B -> 16B-aligned chunks)

#if defined(__HIP_DEVICE_COMPILE__) && __has_builtin(__builtin_amdgcn_global_load_async_to_lds_b128)
#define USE_ASYNC 1
#else
#define USE_ASYNC 0
#endif

#if USE_ASYNC
typedef __attribute__((address_space(1))) v4i g1_v4i;   // global int4
typedef __attribute__((address_space(3))) v4i l3_v4i;   // LDS int4
DEV void async_cp32(__bf16* lds_dst, const __bf16* gsrc) {
  // two B128 async copies: 32 bytes global -> LDS, no VGPR data, tracked by ASYNCcnt
  __builtin_amdgcn_global_load_async_to_lds_b128(
      (g1_v4i*)gsrc, (l3_v4i*)lds_dst, 0, 0);
  __builtin_amdgcn_global_load_async_to_lds_b128(
      (g1_v4i*)(gsrc + 8), (l3_v4i*)(lds_dst + 8), 0, 0);
}
#if __has_builtin(__builtin_amdgcn_s_wait_asynccnt)
#define WAIT_ASYNC(n) __builtin_amdgcn_s_wait_asynccnt(n)
#else
#define WAIT_ASYNC(n) asm volatile("s_wait_asynccnt %0" ::"i"(n) : "memory")
#endif
#endif

DEV float qgelu_f(float v) { return v / (1.f + __expf(-1.702f * v)); }

union V16U { v16bf v; v8bf h[2]; };

// Build a 16x32 bf16 WMMA operand fragment from an LDS row (n/m-major, k-contig).
// Lane layout per ISA 7.12.2: lanes 0-15 hold k = [0..7] & [16..23],
// lanes 16-31 hold k = [8..15] & [24..31] (hi = lane>>4 selects the +8 shift).
DEV v16bf frag_from_lds(const __bf16* rowptr, int hi) {
  V16U u;
  u.h[0] = *(const v8bf*)(rowptr + (hi << 3));       // k = 8*hi .. +8
  u.h[1] = *(const v8bf*)(rowptr + 16 + (hi << 3));  // k = 16+8*hi .. +8
  return u.v;
}

DEV void stage16_f32(__bf16* dst, const float* src) {
#pragma unroll
  for (int q = 0; q < 4; ++q) {
    v4f f = *(const v4f*)(src + 4 * q);
    dst[4 * q + 0] = (__bf16)f[0];
    dst[4 * q + 1] = (__bf16)f[1];
    dst[4 * q + 2] = (__bf16)f[2];
    dst[4 * q + 3] = (__bf16)f[3];
  }
}
DEV void stage16_zero(__bf16* dst) {
#pragma unroll
  for (int i = 0; i < 16; ++i) dst[i] = (__bf16)0.f;
}
DEV void stage16_bf16(__bf16* dst, const __bf16* src) {
  *(v8bf*)dst       = *(const v8bf*)src;
  *(v8bf*)(dst + 8) = *(const v8bf*)(src + 8);
}

// ---------------------------------------------------------------------------
// Kernel 0: repack conv_w (E,Do,Di,3,3) f32 -> (E,9,Do,Di) bf16 (k-contiguous B tiles)
// ---------------------------------------------------------------------------
__global__ __launch_bounds__(256) void vim_repack_conv(const float* __restrict__ cw,
                                                       __bf16* __restrict__ wt) {
  int idx = blockIdx.x * 256 + threadIdx.x;
  const int total = EE * 9 * DD * DD;
  if (idx >= total) return;
  int i = idx & (DD - 1);
  int rest = idx >> 7;
  int o = rest & (DD - 1);
  int rest2 = rest >> 7;
  int t = rest2 % 9;
  int e = rest2 / 9;
  wt[idx] = (__bf16)cw[(((size_t)e * DD + o) * DD + i) * 9 + t];
}

// ---------------------------------------------------------------------------
// Kernel 1: down-projection  xd[e,j,d] = qgelu( X[j,:] . down_w[e,d,:] + down_b[e,d] )
// M=J, N=128, K=1024. A = x (j-major, c contig), B = down_w (d-major, c contig).
// f32 sources must pass through VALU for cvt -> keep sync staging + prefetch.
// ---------------------------------------------------------------------------
__global__ __launch_bounds__(256) void vim_down(const float* __restrict__ x,
                                                const float* __restrict__ dw,
                                                const float* __restrict__ db,
                                                __bf16* __restrict__ xd) {
  __shared__ __bf16 As[128 * LS];
  __shared__ __bf16 Bs[128 * LS];
  const int tid  = threadIdx.x;
  const int lane = tid & 31, wave = tid >> 5;
  const int hi = lane >> 4, ln = lane & 15;
  const int jt0 = blockIdx.x * 128;
  const int e   = blockIdx.y;

  v8f acc[8] = {};

  const int srow  = tid >> 1;
  const int shalf = (tid & 1) << 4;
  const int jrow  = jt0 + srow;
  const float* xsrc = x + (size_t)jrow * CC + shalf;
  const float* wsrc = dw + ((size_t)e * DD + srow) * CC + shalf;
  __bf16* aDst = &As[srow * LS + shalf];
  __bf16* bDst = &Bs[srow * LS + shalf];
  const __bf16* aRow = &As[(wave * 16 + ln) * LS];

  for (int kc = 0; kc < CC / 32; ++kc) {
    const int c0 = kc * 32;
    if (jrow < JJ) stage16_f32(aDst, xsrc + c0); else stage16_zero(aDst);
    stage16_f32(bDst, wsrc + c0);
    __syncthreads();
    v16bf af = frag_from_lds(aRow, hi);
#pragma unroll
    for (int nb = 0; nb < 8; ++nb) {
      v16bf bf = frag_from_lds(&Bs[(nb * 16 + ln) * LS], hi);
      acc[nb] = __builtin_amdgcn_wmma_f32_16x16x32_bf16(
          false, af, false, bf, (short)0, acc[nb], false, false);
    }
    __syncthreads();
    if (kc + 1 < CC / 32 && jrow < JJ)
      __builtin_prefetch(xsrc + c0 + 32, 0, 1);   // global_prefetch_b8
  }

#pragma unroll
  for (int nb = 0; nb < 8; ++nb) {
    const int col = nb * 16 + ln;
    const float bias = db[e * DD + col];
#pragma unroll
    for (int r = 0; r < 8; ++r) {
      int j = jt0 + wave * 16 + r + (hi << 3);
      if (j < JJ)
        xd[((size_t)e * JJ + j) * DD + col] = (__bf16)qgelu_f(acc[nb][r] + bias);
    }
  }
}

// ---------------------------------------------------------------------------
// Kernel 2: 3x3 conv as implicit GEMM. K = 9 taps x 4 chunks of 32 (36 steps).
// cls row (n==0) uses only the center tap; patch rows zero-pad off-grid taps.
// Async double-buffered pipeline:
//   prologue: issue steps 0,1
//   step s:   s_wait_asynccnt<=4 (step s landed; async loads retire in order),
//             zero-patch invalid rows, barrier, WMMA from buf[s&1], barrier,
//             issue step s+2 into buf[s&1] (safe: everyone is done reading it).
// Every thread always issues exactly 4 async B128s per step (dummy row for
// invalid taps) so per-wave ASYNCcnt accounting stays uniform.
// ---------------------------------------------------------------------------
__global__ __launch_bounds__(256) void vim_conv(const __bf16* __restrict__ xd,
                                                const __bf16* __restrict__ wt,
                                                const float* __restrict__ cb,
                                                __bf16* __restrict__ xd2) {
  __shared__ __bf16 As[2][128 * LS];
  __shared__ __bf16 Bs[2][128 * LS];
  const int tid  = threadIdx.x;
  const int lane = tid & 31, wave = tid >> 5;
  const int hi = lane >> 4, ln = lane & 15;
  const int jt0 = blockIdx.x * 128;
  const int e   = blockIdx.y;

  v8f acc[8] = {};

  const int srow   = tid >> 1;
  const int shalf  = (tid & 1) << 4;
  const int j      = jt0 + srow;
  const int ldsoff = srow * LS + shalf;

  // decode this staging row's token coords once
  const int ntok = j >> 5;       // j / B  (B == 32)
  const int b    = j & 31;
  int h = 0, w = 0;
  bool isPatch = false, isCls = false;
  if (j < JJ) {
    if (ntok == 0) isCls = true;
    else { int p = ntok - 1; h = p / 24; w = p % 24; isPatch = true; }
  }

  // neighbor row + validity for tap t (jj -> 0 dummy when invalid)
  auto tapinfo = [&](int t, bool& valid, int& jj) {
    const int dh = t / 3 - 1, dw = t % 3 - 1;
    valid = false; jj = 0;
    if (isPatch) {
      int hh = h + dh, ww = w + dw;
      valid = (hh >= 0) && (hh < 24) && (ww >= 0) && (ww < 24);
      if (valid) jj = ((1 + hh * 24 + ww) << 5) + b;
    } else if (isCls && (t == 4)) {
      valid = true; jj = j;
    }
  };

  auto issue = [&](int s) {
    const int t = s >> 2, kc = s & 3, i0 = kc * 32;
    const int bufi = s & 1;
    bool valid; int jj;
    tapinfo(t, valid, jj);
    const __bf16* asrc = xd + ((size_t)e * JJ + jj) * DD + i0 + shalf;
    const __bf16* bsrc = wt + (((size_t)e * 9 + t) * DD + srow) * DD + i0 + shalf;
#if USE_ASYNC
    async_cp32(&As[bufi][ldsoff], asrc);   // dummy row when invalid, patched later
    async_cp32(&Bs[bufi][ldsoff], bsrc);
#else
    if (valid) stage16_bf16(&As[bufi][ldsoff], asrc);
    else       stage16_zero(&As[bufi][ldsoff]);
    stage16_bf16(&Bs[bufi][ldsoff], bsrc);
#endif
  };

#if USE_ASYNC
  issue(0);
  issue(1);
#endif
  for (int s = 0; s < 36; ++s) {
    const int bufi = s & 1;
#if USE_ASYNC
    if (s == 35) { WAIT_ASYNC(0); } else { WAIT_ASYNC(4); }
    { bool valid; int jj; tapinfo(s >> 2, valid, jj);
      if (!valid) stage16_zero(&As[bufi][ldsoff]); }
#else
    issue(s);
#endif
    __syncthreads();
    v16bf af = frag_from_lds(&As[bufi][(wave * 16 + ln) * LS], hi);
#pragma unroll
    for (int nb = 0; nb < 8; ++nb) {
      v16bf bf = frag_from_lds(&Bs[bufi][(nb * 16 + ln) * LS], hi);
      acc[nb] = __builtin_amdgcn_wmma_f32_16x16x32_bf16(
          false, af, false, bf, (short)0, acc[nb], false, false);
    }
    __syncthreads();
#if USE_ASYNC
    if (s + 2 < 36) issue(s + 2);
#endif
  }

#pragma unroll
  for (int nb = 0; nb < 8; ++nb) {
    const int col = nb * 16 + ln;
    const float bias = cb[e * DD + col];
#pragma unroll
    for (int r = 0; r < 8; ++r) {
      int jr = jt0 + wave * 16 + r + (hi << 3);
      if (jr < JJ)
        xd2[((size_t)e * JJ + jr) * DD + col] = (__bf16)qgelu_f(acc[nb][r] + bias);
    }
  }
}

// ---------------------------------------------------------------------------
// Kernel 3: up-projection, experts accumulated in registers. Gate is folded
// into the B (weight) operand during its f32->bf16 convert, so the xd2 A tile
// is a pure bf16 copy -> async global->LDS.
// out[j, c] = sum_e ( (g_e*up_w[e,c,:]) . xd2[e,j,:] ) + sum_e g_e*up_b[e,c].
// ---------------------------------------------------------------------------
__global__ __launch_bounds__(256) void vim_up(const __bf16* __restrict__ xd2,
                                              const float* __restrict__ uw,
                                              const float* __restrict__ ub,
                                              const float* __restrict__ agg,
                                              float* __restrict__ out) {
  __shared__ __bf16 As[128 * LS];
  __shared__ __bf16 Bs[128 * LS];
  const int tid  = threadIdx.x;
  const int lane = tid & 31, wave = tid >> 5;
  const int hi = lane >> 4, ln = lane & 15;
  const int jt0 = blockIdx.x * 128;
  const int ct  = blockIdx.y;        // 128-col tile of C=1024

  // softmax gates (8 scalar loads, L2-resident)
  float g[EE];
  {
    float mx = agg[0];
#pragma unroll
    for (int i = 1; i < EE; ++i) mx = fmaxf(mx, agg[i]);
    float s = 0.f;
#pragma unroll
    for (int i = 0; i < EE; ++i) { g[i] = __expf(agg[i] - mx); s += g[i]; }
    float inv = 1.f / s;
#pragma unroll
    for (int i = 0; i < EE; ++i) g[i] *= inv;
  }

  v8f acc[8] = {};

  const int srow  = tid >> 1;
  const int shalf = (tid & 1) << 4;
  const int j     = jt0 + srow;
  const int jsafe = (j < JJ) ? j : 0;   // dummy row for OOB staging threads
  __bf16* aDst = &As[srow * LS + shalf];
  __bf16* bDst = &Bs[srow * LS + shalf];
  const __bf16* aRow = &As[(wave * 16 + ln) * LS];

  for (int e = 0; e < EE; ++e) {
    const float ge = g[e];
    const __bf16* asrc = xd2 + ((size_t)e * JJ + jsafe) * DD + shalf;
    const float*  bsrc = uw + ((size_t)e * CC + ct * 128 + srow) * DD + shalf;
#pragma unroll
    for (int kc = 0; kc < 4; ++kc) {
      const int d0 = kc * 32;
#if USE_ASYNC
      async_cp32(aDst, asrc + d0);
#else
      if (j < JJ) stage16_bf16(aDst, asrc + d0); else stage16_zero(aDst);
#endif
      // gate-scaled weight tile (f32 -> bf16 convert is on VALU anyway)
#pragma unroll
      for (int q = 0; q < 4; ++q) {
        v4f f = *(const v4f*)(bsrc + d0 + 4 * q);
        bDst[4 * q + 0] = (__bf16)(ge * f[0]);
        bDst[4 * q + 1] = (__bf16)(ge * f[1]);
        bDst[4 * q + 2] = (__bf16)(ge * f[2]);
        bDst[4 * q + 3] = (__bf16)(ge * f[3]);
      }
#if USE_ASYNC
      WAIT_ASYNC(0);
      if (j >= JJ) stage16_zero(aDst);
#endif
      __syncthreads();
      v16bf af = frag_from_lds(aRow, hi);
#pragma unroll
      for (int nb = 0; nb < 8; ++nb) {
        v16bf bf = frag_from_lds(&Bs[(nb * 16 + ln) * LS], hi);
        acc[nb] = __builtin_amdgcn_wmma_f32_16x16x32_bf16(
            false, af, false, bf, (short)0, acc[nb], false, false);
      }
      __syncthreads();
    }
  }

#pragma unroll
  for (int nb = 0; nb < 8; ++nb) {
    const int colg = ct * 128 + nb * 16 + ln;
    float bias = 0.f;
#pragma unroll
    for (int e = 0; e < EE; ++e) bias += g[e] * ub[e * CC + colg];
#pragma unroll
    for (int r = 0; r < 8; ++r) {
      int jr = jt0 + wave * 16 + r + (hi << 3);
      if (jr < JJ) out[(size_t)jr * CC + colg] = acc[nb][r] + bias;
    }
  }
}

// ---------------------------------------------------------------------------
extern "C" void kernel_launch(void* const* d_in, const int* in_sizes, int n_in,
                              void* d_out, int out_size, void* d_ws, size_t ws_size,
                              hipStream_t stream) {
  const float* x      = (const float*)d_in[0];
  const float* agg    = (const float*)d_in[1];
  const float* down_w = (const float*)d_in[2];
  const float* down_b = (const float*)d_in[3];
  const float* conv_w = (const float*)d_in[4];
  const float* conv_b = (const float*)d_in[5];
  const float* up_w   = (const float*)d_in[6];
  const float* up_b   = (const float*)d_in[7];
  float* out = (float*)d_out;

  // workspace carve-up (bf16 intermediates)
  char* ws = (char*)d_ws;
  const size_t wconv_bytes = (size_t)EE * 9 * DD * DD * 2;      // 2,359,296
  const size_t xd_bytes    = (size_t)EE * JJ * DD * 2;          // 37,814,272
  __bf16* wconv = (__bf16*)ws;
  __bf16* xd    = (__bf16*)(ws + wconv_bytes);
  __bf16* xd2   = (__bf16*)(ws + wconv_bytes + xd_bytes);

  const int repack_total = EE * 9 * DD * DD;
  vim_repack_conv<<<(repack_total + 255) / 256, 256, 0, stream>>>(conv_w, wconv);
  vim_down<<<dim3(JTILES, EE), 256, 0, stream>>>(x, down_w, down_b, xd);
  vim_conv<<<dim3(JTILES, EE), 256, 0, stream>>>(xd, wconv, conv_b, xd2);
  vim_up<<<dim3(JTILES, CC / 128), 256, 0, stream>>>(xd2, up_w, up_b, agg, out);
}
